// SelfAttention_37185826849486
// MI455X (gfx1250) — compile-verified, matrix operand order checked by reading, not run on hardware
//
#include <hip/hip_runtime.h>

// ---------------------------------------------------------------------------
// Types
// ---------------------------------------------------------------------------
typedef unsigned short u16;
typedef __attribute__((ext_vector_type(16))) __bf16 v16bf;
typedef __attribute__((ext_vector_type(8)))  float  v8f;
typedef __attribute__((ext_vector_type(16))) u16    u16x16;
typedef __attribute__((ext_vector_type(8)))  u16    u16x8;
typedef __attribute__((ext_vector_type(4)))  u16    u16x4;
typedef __attribute__((ext_vector_type(4)))  float  f32x4;

#if __has_builtin(__builtin_amdgcn_exp2f)
#define EXP2F(x) __builtin_amdgcn_exp2f(x)
#else
#define EXP2F(x) exp2f(x)
#endif

#define WMMA_BF16(A, B, C)                                                    \
  __builtin_amdgcn_wmma_f32_16x16x32_bf16(false, (A), false, (B), (short)0,   \
                                          (C), false, false)

__device__ __forceinline__ u16 f32_to_bf16(float f) {
  unsigned u = __builtin_bit_cast(unsigned, f);
  u += 0x7FFFu + ((u >> 16) & 1u);
  return (u16)(u >> 16);
}

// Load a 16x32 bf16 A-fragment (or, identically, a 32x16 B-fragment whose
// "column n" is source row n) from a row-major bf16 buffer.
// ISA 7.12.2: lane<16 holds K 0..7 (VGPR0-3) and 16..23 (VGPR4-7);
// lane>=16 holds K 8..15 and 24..31.  => two contiguous 16B loads per lane.
__device__ __forceinline__ v16bf load_frag_rm(const u16* base, int stride,
                                              int lane) {
  const int r    = lane & 15;
  const int koff = (lane >> 4) << 3;  // 0 or 8
  const u16* p   = base + r * stride + koff;
  union { u16x16 v; u16x8 h[2]; } u;
  u.h[0] = *(const u16x8*)(p);
  u.h[1] = *(const u16x8*)(p + 16);
  return __builtin_bit_cast(v16bf, u.v);
}

// ---------------------------------------------------------------------------
// Kernel 0: fp32 -> bf16 conversion (vectorized)
// ---------------------------------------------------------------------------
__global__ __launch_bounds__(256) void cvt_f32_bf16(
    const f32x4* __restrict__ in, u16x4* __restrict__ out, int n4) {
  int i = blockIdx.x * blockDim.x + threadIdx.x;
  const int stride = gridDim.x * blockDim.x;
  for (; i < n4; i += stride) {
    f32x4 v = in[i];
    u16x4 o;
#pragma unroll
    for (int r = 0; r < 4; ++r) o[r] = f32_to_bf16(v[r]);
    out[i] = o;
  }
}

// ---------------------------------------------------------------------------
// Kernel 1: GEMM  D[M,1024] = A[M,1024] @ W[1024,1024]^T  (bf16 in, wmma f32)
//   MODE 0: bf16 out, row-major, scaled by alpha
//   MODE 1: bf16 out, per-head transposed Vt[(b*1024+n)][2048]
//   MODE 2: fp32 out + bias (final projection into d_out)
// Block = 256 threads (8 waves); block tile 256x64; wave tile 32x64:
// each B (weight) fragment is reused by two WMMAs (rows 0-15 / 16-31).
// ---------------------------------------------------------------------------
template <int MODE>
__global__ __launch_bounds__(256) void gemm_bf16_wmma(
    const u16* __restrict__ A, const u16* __restrict__ W,
    void* __restrict__ out, const float* __restrict__ bias, float alpha) {
  const int lane = threadIdx.x & 31;
  const int wid  = threadIdx.x >> 5;
  const int row0 = blockIdx.x * 256 + wid * 32;
  const int col0 = blockIdx.y * 64;

  v8f acc[8] = {};
  const u16* Arow = A + row0 * 1024;

  for (int k0 = 0; k0 < 1024; k0 += 32) {
    v16bf af0 = load_frag_rm(Arow + k0, 1024, lane);              // rows 0-15
    v16bf af1 = load_frag_rm(Arow + 16 * 1024 + k0, 1024, lane);  // rows 16-31
#pragma unroll
    for (int t = 0; t < 4; ++t) {
      v16bf bf = load_frag_rm(W + (col0 + t * 16) * 1024 + k0, 1024, lane);
      acc[t]     = WMMA_BF16(af0, bf, acc[t]);
      acc[4 + t] = WMMA_BF16(af1, bf, acc[4 + t]);
    }
  }

  const int hl = lane >> 4, nl = lane & 15;
#pragma unroll
  for (int p = 0; p < 2; ++p) {
    const int rbase = row0 + p * 16;
    if constexpr (MODE == 0) {
      u16* O = (u16*)out;
#pragma unroll
      for (int t = 0; t < 4; ++t)
#pragma unroll
        for (int r = 0; r < 8; ++r)
          O[(size_t)(rbase + r + 8 * hl) * 1024 + (col0 + t * 16 + nl)] =
              f32_to_bf16(acc[p * 4 + t][r] * alpha);
    } else if constexpr (MODE == 1) {
      u16* O = (u16*)out;
      const int bb  = rbase >> 11;            // batch (2048 tokens per batch)
      const int tok = (rbase & 2047) + 8 * hl;
#pragma unroll
      for (int t = 0; t < 4; ++t) {
        const int n = col0 + t * 16 + nl;     // output feature = head*64 + dim
        u16x8 pk;
#pragma unroll
        for (int r = 0; r < 8; ++r) pk[r] = f32_to_bf16(acc[p * 4 + t][r] * alpha);
        *(u16x8*)(O + ((size_t)(bb * 1024 + n)) * 2048 + tok) = pk;
      }
    } else {
      float* O = (float*)out;
#pragma unroll
      for (int t = 0; t < 4; ++t) {
        const float bv = bias[col0 + t * 16 + nl];
#pragma unroll
        for (int r = 0; r < 8; ++r)
          O[(size_t)(rbase + r + 8 * hl) * 1024 + (col0 + t * 16 + nl)] =
              acc[p * 4 + t][r] + bv;
      }
    }
  }
}

// ---------------------------------------------------------------------------
// Kernel 2: flash attention.  One wave = 16 query rows of one (batch, head).
// Q is pre-scaled by 0.125*log2(e), so softmax uses exp2 (v_exp_f32).
// Score C-layout: VGPR r, lane l  ->  row r+8*(l/16), col l%16.
// Row max: __shfl_xor butterfly over {1,2,4,8} (stays inside a 16-lane half).
// Row SUM: computed on the matrix pipe as an extra  P @ ones  accumulator
// (its rescale recurrence  l = l*fsc + rowsum  is exactly the flash rescale),
// removing half the ds_bpermute traffic from the inner loop.
// P (C-layout) -> A-layout via 1KB per-wave LDS bounce.
// ---------------------------------------------------------------------------
__global__ __launch_bounds__(256) void flash_attn_wmma(
    const u16* __restrict__ Q, const u16* __restrict__ K,
    const u16* __restrict__ Vt, u16* __restrict__ AO) {
  __shared__ u16 pTile[8][16 * 32];

  const int lane  = threadIdx.x & 31;
  const int wslot = threadIdx.x >> 5;
  const int task  = blockIdx.x * 8 + wslot;   // 0..8191
  const int rt    = task & 127;               // row tile within (b,h)
  const int h     = (task >> 7) & 15;
  const int b     = task >> 11;
  const int tok0  = rt * 16;
  const int hl    = lane >> 4, nl = lane & 15;

  const u16* Qbase = Q + ((b * 2048 + tok0) * 1024 + h * 64);
  const v16bf qa0 = load_frag_rm(Qbase, 1024, lane);       // q-dims  0..31
  const v16bf qa1 = load_frag_rm(Qbase + 32, 1024, lane);  // q-dims 32..63

  // all-ones bf16 B fragment (layout-independent constant)
  u16x16 ov;
#pragma unroll
  for (int i = 0; i < 16; ++i) ov[i] = 0x3F80;  // bf16(1.0)
  const v16bf ones = __builtin_bit_cast(v16bf, ov);

  const u16* Kh = K + (b * 2048 * 1024 + h * 64);
  const u16* Vh = Vt + ((size_t)(b * 1024 + h * 64)) * 2048;
  u16* pt = &pTile[wslot][0];

  v8f   Oa[4] = {};
  v8f   Os    = {};          // running, un-normalized softmax denominator
  float mrow[8];
#pragma unroll
  for (int r = 0; r < 8; ++r) mrow[r] = -1e30f;

  for (int j = 0; j < 2048; j += 32) {
    const u16* Kb0 = Kh + j * 1024;
    // locality=3 -> WGP-scope prefetch (pulls into all cache levels, §10.5);
    // SYS-scope (locality 0) would only fill GL2, which is already hot.
    __builtin_prefetch(Kb0 + 32 * 1024, 0, 3);
    __builtin_prefetch(Vh + j + 32, 0, 3);

    // S tile 16x32 (two 16x16 accumulators), already in 2^x scale
    v8f s0 = {}, s1 = {};
    s0 = WMMA_BF16(qa0, load_frag_rm(Kb0, 1024, lane), s0);
    s0 = WMMA_BF16(qa1, load_frag_rm(Kb0 + 32, 1024, lane), s0);
    s1 = WMMA_BF16(qa0, load_frag_rm(Kb0 + 16 * 1024, 1024, lane), s1);
    s1 = WMMA_BF16(qa1, load_frag_rm(Kb0 + 16 * 1024 + 32, 1024, lane), s1);

    // ---- online softmax: row max butterfly (16-lane halves) ----
    float mx[8];
#pragma unroll
    for (int r = 0; r < 8; ++r) mx[r] = fmaxf(s0[r], s1[r]);
#pragma unroll
    for (int off = 1; off < 16; off <<= 1)
#pragma unroll
      for (int r = 0; r < 8; ++r)
        mx[r] = fmaxf(mx[r], __shfl_xor(mx[r], off, 32));

    float fsc[8];
#pragma unroll
    for (int r = 0; r < 8; ++r) {
      const float mn = fmaxf(mrow[r], mx[r]);
      fsc[r]  = EXP2F(mrow[r] - mn);
      mrow[r] = mn;
      s0[r]   = EXP2F(s0[r] - mn);
      s1[r]   = EXP2F(s1[r] - mn);
    }
    // rescale running accumulators (incl. denominator accumulator)
#pragma unroll
    for (int r = 0; r < 8; ++r) Os[r] *= fsc[r];
#pragma unroll
    for (int t = 0; t < 4; ++t)
#pragma unroll
      for (int r = 0; r < 8; ++r) Oa[t][r] *= fsc[r];

    // ---- P: C-layout -> A-layout via per-wave LDS tile (16x32 bf16) ----
#pragma unroll
    for (int r = 0; r < 8; ++r) {
      pt[(r + 8 * hl) * 32 + nl]      = f32_to_bf16(s0[r]);
      pt[(r + 8 * hl) * 32 + 16 + nl] = f32_to_bf16(s1[r]);
    }
    asm volatile("" ::: "memory");  // LDS is in-order within a wave
    const v16bf pa = load_frag_rm(pt, 32, lane);
    asm volatile("" ::: "memory");

    // ---- O += P @ V ;  Os += P @ ones (row sums on the matrix pipe) ----
    Os = WMMA_BF16(pa, ones, Os);
#pragma unroll
    for (int t = 0; t < 4; ++t) {
      v16bf vb = load_frag_rm(Vh + (t * 16) * 2048 + j, 2048, lane);
      Oa[t] = WMMA_BF16(pa, vb, Oa[t]);
    }
  }

  // epilogue: normalize by Os (every column holds the row sum) and store bf16
  float inv[8];
#pragma unroll
  for (int r = 0; r < 8; ++r) inv[r] = 1.0f / Os[r];
  u16* aobase = AO + ((b * 2048 + tok0) * 1024 + h * 64);
#pragma unroll
  for (int t = 0; t < 4; ++t)
#pragma unroll
    for (int r = 0; r < 8; ++r)
      aobase[(size_t)(r + 8 * hl) * 1024 + t * 16 + nl] =
          f32_to_bf16(Oa[t][r] * inv[r]);
}

// ---------------------------------------------------------------------------
// Launch
// ---------------------------------------------------------------------------
extern "C" void kernel_launch(void* const* d_in, const int* in_sizes, int n_in,
                              void* d_out, int out_size, void* d_ws,
                              size_t ws_size, hipStream_t stream) {
  const float* x  = (const float*)d_in[0];
  const float* Wk = (const float*)d_in[1];
  const float* Wq = (const float*)d_in[2];
  const float* Wv = (const float*)d_in[3];
  const float* Wu = (const float*)d_in[4];
  const float* bu = (const float*)d_in[5];
  float* out = (float*)d_out;

  char* ws = (char*)d_ws;
  u16* xb  = (u16*)(ws + ((size_t)0 << 20));   // 16 MB  x bf16
  u16* wkb = (u16*)(ws + ((size_t)16 << 20));  //  2 MB
  u16* wqb = (u16*)(ws + ((size_t)18 << 20));
  u16* wvb = (u16*)(ws + ((size_t)20 << 20));
  u16* wub = (u16*)(ws + ((size_t)22 << 20));
  u16* Qb  = (u16*)(ws + ((size_t)24 << 20));  // 16 MB each
  u16* Kb  = (u16*)(ws + ((size_t)40 << 20));
  u16* Vt  = (u16*)(ws + ((size_t)56 << 20));
  u16* AO  = (u16*)(ws + ((size_t)72 << 20));

  // 1) convert inputs to bf16
  cvt_f32_bf16<<<1024, 256, 0, stream>>>((const f32x4*)x, (u16x4*)xb,
                                         8388608 / 4);
  cvt_f32_bf16<<<256, 256, 0, stream>>>((const f32x4*)Wk, (u16x4*)wkb,
                                        1048576 / 4);
  cvt_f32_bf16<<<256, 256, 0, stream>>>((const f32x4*)Wq, (u16x4*)wqb,
                                        1048576 / 4);
  cvt_f32_bf16<<<256, 256, 0, stream>>>((const f32x4*)Wv, (u16x4*)wvb,
                                        1048576 / 4);
  cvt_f32_bf16<<<256, 256, 0, stream>>>((const f32x4*)Wu, (u16x4*)wub,
                                        1048576 / 4);

  // 2) projections (M=8192).  Q carries 1/sqrt(64) * log2(e) for exp2 softmax.
  const dim3 ggrid(32, 16);  // 256-row x 64-col block tiles
  const float qscale = 0.125f * 1.44269504088896340736f;
  gemm_bf16_wmma<0><<<ggrid, 256, 0, stream>>>(xb, wqb, Qb, nullptr, qscale);
  gemm_bf16_wmma<0><<<ggrid, 256, 0, stream>>>(xb, wkb, Kb, nullptr, 1.0f);
  gemm_bf16_wmma<1><<<ggrid, 256, 0, stream>>>(xb, wvb, Vt, nullptr, 1.0f);

  // 3) flash attention: 8192 wave-tasks, 8 waves / block
  flash_attn_wmma<<<1024, 256, 0, stream>>>(Qb, Kb, Vt, AO);

  // 4) output projection + bias -> fp32 d_out
  gemm_bf16_wmma<2><<<ggrid, 256, 0, stream>>>(AO, wub, out, bu, 1.0f);
}
